// GraphIsomorphismNetwork_82952998355943
// MI455X (gfx1250) — matrix operand order repaired and checked by reading, not compiled, not run on hardware
//
#include <hip/hip_runtime.h>
#include <hip/hip_bf16.h>

// GIN inference: N=20000 nodes, 320000 edges, D_IN=D_HID=256, D_OUT=128, 5 layers.
// bf16 WMMA (v_wmma_f32_16x16x32_bf16), TDM-staged weights in LDS, fp32 accumulation.

typedef __attribute__((ext_vector_type(16))) __bf16 v16bf;
typedef __attribute__((ext_vector_type(8)))  __bf16 v8bf;
typedef __attribute__((ext_vector_type(8)))  float  v8f;
typedef __attribute__((ext_vector_type(4)))  unsigned int v4u;
typedef __attribute__((ext_vector_type(8)))  int v8i;
typedef __attribute__((ext_vector_type(4)))  int v4i;

union AFrag { v16bf v; struct { v8bf lo; v8bf hi; } p; };

__device__ __forceinline__ unsigned short f2bf(float f) {
    unsigned int u = __float_as_uint(f);
    u += 0x7FFFu + ((u >> 16) & 1u);        // round-to-nearest-even
    return (unsigned short)(u >> 16);
}

// ---------------- agg init: agg = (1+eps)*h ----------------
__global__ void init_agg(const float* __restrict__ h, const float* __restrict__ eps,
                         int li, float* __restrict__ agg, int n) {
    int i = blockIdx.x * 256 + threadIdx.x;
    if (i < n) agg[i] = (1.0f + eps[li]) * h[i];
}

// ---------------- weight pre-swizzle into WMMA B-fragment layout ----------------
// Source: W[k][col] row-major fp32, K x N.  Dest linear index:
//   (((ct*nk + ks)*32 + lane)*16 + e)   where col = ct*16 + (lane&15),
//   k = ks*32 + (lane>=16 ? 16 : 0) + e
__global__ void pack_w(const float* __restrict__ w, unsigned short* __restrict__ dst,
                       int K, int N) {
    int idx = blockIdx.x * 256 + threadIdx.x;
    int total = K * N;
    if (idx >= total) return;
    int e    = idx & 15;
    int lane = (idx >> 4) & 31;
    int t    = idx >> 9;
    int nk   = K >> 5;
    int ks   = t % nk;
    int ct   = t / nk;
    int col  = (ct << 4) + (lane & 15);
    int k    = (ks << 5) + ((lane >> 4) << 4) + e;
    dst[idx] = f2bf(w[k * N + col]);
}

// ---------------- edge scatter-add: agg[dst] += h[src] ----------------
__global__ void scatter_add(const float* __restrict__ h, const int* __restrict__ src,
                            const int* __restrict__ dst, float* __restrict__ agg,
                            int nedges) {
    int tid = blockIdx.x * 256 + threadIdx.x;
    int e = tid >> 6;
    if (e >= nedges) return;
    int d = (tid & 63) << 2;
    int s = src[e];
    int t = dst[e];
    const float4 v = *(const float4*)(h + s * 256 + d);
    float* a = agg + t * 256 + d;
    __hip_atomic_fetch_add(a + 0, v.x, __ATOMIC_RELAXED, __HIP_MEMORY_SCOPE_AGENT);
    __hip_atomic_fetch_add(a + 1, v.y, __ATOMIC_RELAXED, __HIP_MEMORY_SCOPE_AGENT);
    __hip_atomic_fetch_add(a + 2, v.z, __ATOMIC_RELAXED, __HIP_MEMORY_SCOPE_AGENT);
    __hip_atomic_fetch_add(a + 3, v.w, __ATOMIC_RELAXED, __HIP_MEMORY_SCOPE_AGENT);
}

// ---------------- z = bf16(agg) ----------------
__global__ void cvt_bf16(const float* __restrict__ agg, unsigned short* __restrict__ z, int n) {
    int i = blockIdx.x * 256 + threadIdx.x;
    if (i < n) z[i] = f2bf(agg[i]);
}

// ---------------- fused GEMM: out = epilogue(A[Mx256] @ B[256xN] + bias) ----------------
// Block = 64 rows x N cols, 8 waves: wave w -> row-tile (w>>1), col-half (w&1).
// Packed B (nct*8KB) is TDM-copied into LDS once, WMMAs read B via ds_load.
// mode 0: ReLU -> bf16 | mode 1: BN+ReLU -> f32 | mode 2: plain -> f32
__global__ __launch_bounds__(256) void gin_gemm(
    const unsigned short* __restrict__ A, const unsigned short* __restrict__ Bp,
    const float* __restrict__ bias, int M, int N, int mode,
    unsigned short* __restrict__ out_bf, float* __restrict__ out_f,
    const float* __restrict__ g, const float* __restrict__ bt,
    const float* __restrict__ mu, const float* __restrict__ var) {
    extern __shared__ unsigned short bsm[];   // nct*8192 ushorts (256KB for N=256)

    const int wave = threadIdx.x >> 5;
    const int lane = threadIdx.x & 31;
    const int lrow = lane & 15;
    const int ksel = lane >> 4;              // 0: lanes 0-15, 1: lanes 16-31
    const int nct  = N >> 4;
    const int rt   = wave >> 1;              // row tile 0..3
    const int ch   = wave & 1;               // column half
    const int tpw  = nct >> 1;               // col tiles per wave
    const int row0 = blockIdx.x * 64 + rt * 16;

    // --- TDM: bulk-copy packed B (linear bytes) into LDS, wave 0 only ---
    if (wave == 0) {
        const unsigned long long ga = (unsigned long long)(const void*)Bp;
        const unsigned int nelem8 = (unsigned int)(nct * 1024);   // 8-byte elements
        v4u g0;
        g0[0] = 1u;                                         // count=1, user mode
        g0[1] = 0u;                                         // lds_addr = 0 (dyn-LDS base)
        g0[2] = (unsigned int)(ga & 0xFFFFFFFFu);           // global_addr[31:0]
        g0[3] = (unsigned int)((ga >> 32) & 0x1FFFFFFu)     // global_addr[56:32]
              | (2u << 30);                                 // type = 2 ("image")
        v8i g1;
        g1[0] = (int)(3u << 16);                            // data_size = 8B
        g1[1] = (int)((nelem8 & 0xFFFFu) << 16);            // tensor_dim0[15:0]
        g1[2] = (int)((nelem8 >> 16) & 0xFFFFu);            // tensor_dim0[31:16], tensor_dim1=0
        g1[3] = (int)((nelem8 & 0xFFFFu) << 16);            // tile_dim0 = nelem8 (1D tile)
        g1[4] = 0;                                          // tile_dim1/2 = 0 (unused)
        g1[5] = (int)nelem8;                                // tensor_dim0_stride[31:0]
        g1[6] = 0;
        g1[7] = 0;
        v4i gz4 = {0, 0, 0, 0};
        v8i gz8 = {0, 0, 0, 0, 0, 0, 0, 0};
        __builtin_amdgcn_tensor_load_to_lds(g0, g1, gz4, gz4, gz8, 0);
        __builtin_amdgcn_s_wait_tensorcnt(0);
    }

    // Preload A fragments for this wave's row tile (K=256 -> 8 slabs).
    int row = row0 + lrow;
    if (row >= M) row = M - 1;
    AFrag af[8];
    const unsigned short* arow = A + (size_t)row * 256;
    #pragma unroll
    for (int ks = 0; ks < 8; ++ks) {
        af[ks].p.lo = *(const v8bf*)(arow + ks * 32 + ksel * 16);
        af[ks].p.hi = *(const v8bf*)(arow + ks * 32 + ksel * 16 + 8);
    }

    __syncthreads();   // LDS weights visible to all waves

    for (int t = 0; t < tpw; ++t) {
        const int ct = ch * tpw + t;
        const unsigned short* bbase = bsm + ((size_t)(ct * 8) * 32 + lane) * 16;

        v8f acc = {};
        #pragma unroll
        for (int ks = 0; ks < 8; ++ks) {
            v16bf bfr = *(const v16bf*)(bbase + ks * 512);
            acc = __builtin_amdgcn_wmma_f32_16x16x32_bf16(
                false, af[ks].v, false, bfr, (short)0, acc, false, false);
        }

        // C/D layout: lane -> col = ct*16 + (lane&15); VGPR j -> row j (+8 for lanes>=16)
        const int col = (ct << 4) + lrow;
        const int mb  = row0 + (ksel << 3);
        const float bv = bias[col];
        if (mode == 0) {
            #pragma unroll
            for (int j = 0; j < 8; ++j) {
                int r = mb + j;
                if (r < M) {
                    float v = acc[j] + bv;
                    v = v > 0.0f ? v : 0.0f;
                    out_bf[(size_t)r * N + col] = f2bf(v);
                }
            }
        } else if (mode == 1) {
            const float sc = g[col] * rsqrtf(var[col] + 1e-5f);
            const float sh = bt[col] - mu[col] * sc;
            #pragma unroll
            for (int j = 0; j < 8; ++j) {
                int r = mb + j;
                if (r < M) {
                    float v = (acc[j] + bv) * sc + sh;
                    v = v > 0.0f ? v : 0.0f;
                    out_f[(size_t)r * N + col] = v;
                }
            }
        } else {
            #pragma unroll
            for (int j = 0; j < 8; ++j) {
                int r = mb + j;
                if (r < M) out_f[(size_t)r * N + col] = acc[j] + bv;
            }
        }
    }
}

extern "C" void kernel_launch(void* const* d_in, const int* in_sizes, int n_in,
                              void* d_out, int out_size, void* d_ws, size_t ws_size,
                              hipStream_t stream) {
    const float* x   = (const float*)d_in[0];
    const int*   ei  = (const int*)d_in[1];
    const float* W1  = (const float*)d_in[2];
    const float* b1  = (const float*)d_in[3];
    const float* W2a = (const float*)d_in[4];
    const float* b2a = (const float*)d_in[5];
    const float* WL  = (const float*)d_in[6];
    const float* bL  = (const float*)d_in[7];
    const float* eps = (const float*)d_in[8];
    const float* gam = (const float*)d_in[9];
    const float* bet = (const float*)d_in[10];
    const float* mu  = (const float*)d_in[11];
    const float* var = (const float*)d_in[12];

    const int NN = in_sizes[0] / 256;   // 20000 nodes
    const int NE = in_sizes[1] / 2;     // 320000 edges
    const int NF = NN * 256;

    // Carve workspace (256B aligned)
    char* w = (char*)d_ws;
    auto carve = [&](size_t bytes) -> void* {
        void* p = (void*)w;
        w += (bytes + 255) & ~(size_t)255;
        return p;
    };
    float*          hbuf = (float*)carve((size_t)NF * 4);
    float*          agg  = (float*)carve((size_t)NF * 4);
    unsigned short* zbuf = (unsigned short*)carve((size_t)NF * 2);
    unsigned short* ybuf = (unsigned short*)carve((size_t)NF * 2);
    unsigned short* W1p  = (unsigned short*)carve((size_t)5 * 65536 * 2);
    unsigned short* W2p  = (unsigned short*)carve((size_t)4 * 65536 * 2);
    unsigned short* WLp  = (unsigned short*)carve((size_t)32768 * 2);

    // Pre-swizzle weights into WMMA B-fragment layout (bf16)
    for (int i = 0; i < 5; ++i)
        pack_w<<<(65536 + 255) / 256, 256, 0, stream>>>(W1 + i * 65536, W1p + i * 65536, 256, 256);
    for (int i = 0; i < 4; ++i)
        pack_w<<<(65536 + 255) / 256, 256, 0, stream>>>(W2a + i * 65536, W2p + i * 65536, 256, 256);
    pack_w<<<(32768 + 255) / 256, 256, 0, stream>>>(WL, WLp, 256, 128);

    const int elemGrid = (NF + 255) / 256;
    const int scatGrid = (NE * 64 + 255) / 256;
    const int gemmGrid = (NN + 63) / 64;
    const size_t smemFull = 16 * 8192 * sizeof(unsigned short);  // 256KB (N=256)
    const size_t smemHalf = 8 * 8192 * sizeof(unsigned short);   // 128KB (N=128)

    for (int i = 0; i < 5; ++i) {
        const float* hin = (i == 0) ? x : hbuf;
        // agg = (1+eps_i)*h, then scatter neighbor sums on top, then convert to bf16
        init_agg<<<elemGrid, 256, 0, stream>>>(hin, eps, i, agg, NF);
        scatter_add<<<scatGrid, 256, 0, stream>>>(hin, ei, ei + NE, agg, NE);
        cvt_bf16<<<elemGrid, 256, 0, stream>>>(agg, zbuf, NF);
        // MLP layer 1: y = relu(z @ W1[i] + b1[i]) -> bf16
        gin_gemm<<<gemmGrid, 256, smemFull, stream>>>(zbuf, W1p + i * 65536, b1 + i * 256,
                                                      NN, 256, 0, ybuf, nullptr,
                                                      nullptr, nullptr, nullptr, nullptr);
        if (i < 4) {
            gin_gemm<<<gemmGrid, 256, smemFull, stream>>>(ybuf, W2p + i * 65536, b2a + i * 256,
                                                          NN, 256, 1, nullptr, hbuf,
                                                          gam + i * 256, bet + i * 256,
                                                          mu + i * 256, var + i * 256);
        } else {
            gin_gemm<<<gemmGrid, 256, smemHalf, stream>>>(ybuf, WLp, bL,
                                                          NN, 128, 2, nullptr, (float*)d_out,
                                                          nullptr, nullptr, nullptr, nullptr);
        }
    }
}
